// AttnPowerP2_47691316855178
// MI455X (gfx1250) — matrix operand order repaired and checked by reading, not compiled, not run on hardware
//
#include <hip/hip_runtime.h>
#include <hip/hip_bf16.h>
#include <stdint.h>

typedef __attribute__((ext_vector_type(16))) __bf16 v16bf;
typedef __attribute__((ext_vector_type(8)))  __bf16 v8bf;
typedef __attribute__((ext_vector_type(4)))  __bf16 v4bf;
typedef __attribute__((ext_vector_type(8)))  float  v8f;
typedef __attribute__((ext_vector_type(4)))  float  v4f;

#define TSEQ   1024
#define DHEAD  64
#define SROW   1032          // padded bf16 row stride (avoids 16-way LDS bank conflicts)
#define EPSV   1e-6f
#define SCALEV 0.125f        // 1/sqrt(64)

__device__ __forceinline__ void put4(v16bf& d, int base, v4f s) {
  d[base + 0] = (__bf16)s.x; d[base + 1] = (__bf16)s.y;
  d[base + 2] = (__bf16)s.z; d[base + 3] = (__bf16)s.w;
}

__global__ __launch_bounds__(128, 1)
void attn_p2_wmma_kernel(const float* __restrict__ q,
                         const float* __restrict__ k,
                         const float* __restrict__ v,
                         const int*   __restrict__ amask,
                         float* __restrict__ out_o,
                         float* __restrict__ p_o)
{
  extern __shared__ char smem[];
  __bf16* Sp     = (__bf16*)smem;                            // [64][SROW]: s^2 then p (bf16)
  __bf16* Vt     = (__bf16*)(smem + 64 * SROW * 2);          // [64 d][SROW k]: V^T bf16
  float*  invden = (float*)(smem + 2 * 64 * SROW * 2);       // [64]

  const int tid  = threadIdx.x;
  const int wid  = tid >> 5;        // wave id 0..3
  const int lane = tid & 31;
  const int n    = lane & 15;       // column (N) / A-row index
  const int h    = lane >> 4;       // lane-half per WMMA layouts

  const int qtile = blockIdx.x & 15;
  const int bh    = blockIdx.x >> 4;
  const int bidx  = bh / 12;
  const int qbase = qtile * 64;

  const float* Q  = q + (size_t)bh * TSEQ * DHEAD;
  const float* K  = k + (size_t)bh * TSEQ * DHEAD;
  const float* V  = v + (size_t)bh * TSEQ * DHEAD;
  const int*   Mk = amask + (size_t)bidx * TSEQ;

  // ---- stage V^T into LDS as bf16 (once per workgroup) ----
  for (int ch = tid; ch < TSEQ / 4; ch += 128) {
    const int r0 = ch << 2;                      // 4 consecutive k rows
    const float* vr = V + (size_t)r0 * DHEAD;
    for (int d0 = 0; d0 < DHEAD; d0 += 4) {
      v4f a0 = *(const v4f*)(vr + d0);
      v4f a1 = *(const v4f*)(vr + DHEAD + d0);
      v4f a2 = *(const v4f*)(vr + 2 * DHEAD + d0);
      v4f a3 = *(const v4f*)(vr + 3 * DHEAD + d0);
      #pragma unroll
      for (int j = 0; j < 4; ++j) {
        v4bf pk;
        pk[0] = (__bf16)a0[j]; pk[1] = (__bf16)a1[j];
        pk[2] = (__bf16)a2[j]; pk[3] = (__bf16)a3[j];
        *(v4bf*)(Vt + (size_t)(d0 + j) * SROW + r0) = pk;   // ds_store_b64
      }
    }
  }
  __syncthreads();

  // ---- Q A-fragments (16-bit A layout: e<8 -> d=8h+e ; e>=8 -> d=16+8h+(e-8)) ----
  const float* qr = Q + (size_t)(qbase + wid * 16 + n) * DHEAD;
  v16bf aq0, aq1;
  put4(aq0, 0,  *(const v4f*)(qr + 8 * h));
  put4(aq0, 4,  *(const v4f*)(qr + 8 * h + 4));
  put4(aq0, 8,  *(const v4f*)(qr + 16 + 8 * h));
  put4(aq0, 12, *(const v4f*)(qr + 16 + 8 * h + 4));
  put4(aq1, 0,  *(const v4f*)(qr + 32 + 8 * h));
  put4(aq1, 4,  *(const v4f*)(qr + 32 + 8 * h + 4));
  put4(aq1, 8,  *(const v4f*)(qr + 48 + 8 * h));
  put4(aq1, 12, *(const v4f*)(qr + 48 + 8 * h + 4));

  // ---- phase 1: s2 = (QK^T*scale + addmask)^2, row sums ----
  float dsum[8] = {0, 0, 0, 0, 0, 0, 0, 0};
  for (int t = 0; t < 64; ++t) {
    const int kr = t * 16 + n;                   // key position for this lane's column
    const float* krow = K + (size_t)kr * DHEAD;  // B[e][n]: e<8 -> d=8h+e ; e>=8 -> d=16+8h+(e-8)
    v16bf bk0, bk1;
    put4(bk0, 0,  *(const v4f*)(krow + 8 * h));
    put4(bk0, 4,  *(const v4f*)(krow + 8 * h + 4));
    put4(bk0, 8,  *(const v4f*)(krow + 16 + 8 * h));
    put4(bk0, 12, *(const v4f*)(krow + 16 + 8 * h + 4));
    put4(bk1, 0,  *(const v4f*)(krow + 32 + 8 * h));
    put4(bk1, 4,  *(const v4f*)(krow + 32 + 8 * h + 4));
    put4(bk1, 8,  *(const v4f*)(krow + 48 + 8 * h));
    put4(bk1, 12, *(const v4f*)(krow + 48 + 8 * h + 4));

    v8f acc = {};
    acc = __builtin_amdgcn_wmma_f32_16x16x32_bf16(false, aq0, false, bk0, (short)0, acc, false, false);
    acc = __builtin_amdgcn_wmma_f32_16x16x32_bf16(false, aq1, false, bk1, (short)0, acc, false, false);

    const float am = Mk[kr] ? 0.0f : -10000.0f;
    __bf16* scol = Sp + (size_t)(wid * 16 + 8 * h) * SROW + kr;
    #pragma unroll
    for (int i = 0; i < 8; ++i) {                // C/D layout: row = i + 8h, col = n
      float s  = acc[i] * SCALEV + am;
      float s2 = s * s;
      dsum[i] += s2;
      scol[(size_t)i * SROW] = (__bf16)s2;
    }
  }

  // ---- row-sum reduction within each 16-lane half (wave32 shuffles) ----
  #pragma unroll
  for (int i = 0; i < 8; ++i) {
    float s = dsum[i];
    s += __shfl_xor(s, 1, 32);
    s += __shfl_xor(s, 2, 32);
    s += __shfl_xor(s, 4, 32);
    s += __shfl_xor(s, 8, 32);
    if (n == 0) invden[wid * 16 + 8 * h + i] = 1.0f / (s + EPSV);
  }
  __syncthreads();

  // ---- normalize: coalesced fp32 p to global, bf16 p back to LDS ----
  for (int r = 0; r < 16; ++r) {
    const int rl = wid * 16 + r;
    const float inv = invden[rl];
    float*  prow = p_o + ((size_t)bh * TSEQ + qbase + rl) * TSEQ;
    __bf16* srow = Sp + (size_t)rl * SROW;
    for (int c = lane * 4; c < TSEQ; c += 128) {
      v4bf sv = *(const v4bf*)(srow + c);
      v4f  pv;
      #pragma unroll
      for (int j = 0; j < 4; ++j) pv[j] = (float)sv[j] * inv;
      *(v4f*)(prow + c) = pv;                    // global_store_b128, coalesced
      v4bf pb;
      #pragma unroll
      for (int j = 0; j < 4; ++j) pb[j] = (__bf16)pv[j];
      *(v4bf*)(srow + c) = pb;
    }
  }

  // ---- phase 2: out = P @ V over 32 k-windows, 4 d-tiles ----
  v8f co[4] = {};
  const __bf16* arow = Sp + (size_t)(wid * 16 + n) * SROW;   // A row = lane&15
  for (int w = 0; w < 32; ++w) {
    const int kb = w * 32;
    v8bf lo = *(const v8bf*)(arow + kb + 8 * h);             // ds_load_b128
    v8bf hi = *(const v8bf*)(arow + kb + 16 + 8 * h);
    v16bf ap;
    #pragma unroll
    for (int e = 0; e < 8; ++e) { ap[e] = lo[e]; ap[8 + e] = hi[e]; }
    #pragma unroll
    for (int dt = 0; dt < 4; ++dt) {
      const __bf16* vtp = Vt + (size_t)(dt * 16 + n) * SROW + kb + 8 * h;
      v8bf b0 = *(const v8bf*)(vtp);                         // B[e][n] = V^T[d][kb+8h+e], e>=8: kb+16+8h+(e-8)
      v8bf b1 = *(const v8bf*)(vtp + 16);
      v16bf bv;
      #pragma unroll
      for (int e = 0; e < 8; ++e) { bv[e] = b0[e]; bv[8 + e] = b1[e]; }
      co[dt] = __builtin_amdgcn_wmma_f32_16x16x32_bf16(false, ap, false, bv, (short)0, co[dt], false, false);
    }
  }

  // ---- write out (C/D layout: row = i + 8h, col = dt*16 + n) ----
  float* ob = out_o + ((size_t)bh * TSEQ + qbase + wid * 16) * DHEAD;
  #pragma unroll
  for (int dt = 0; dt < 4; ++dt) {
    #pragma unroll
    for (int i = 0; i < 8; ++i)
      ob[(size_t)(8 * h + i) * DHEAD + dt * 16 + n] = co[dt][i];
  }
}

extern "C" void kernel_launch(void* const* d_in, const int* in_sizes, int n_in,
                              void* d_out, int out_size, void* d_ws, size_t ws_size,
                              hipStream_t stream) {
  const float* q  = (const float*)d_in[0];
  const float* k  = (const float*)d_in[1];
  const float* v  = (const float*)d_in[2];
  const int*   am = (const int*)d_in[3];

  float* out_o = (float*)d_out;
  float* p_o   = out_o + (size_t)8 * 12 * 1024 * 64;   // tuple: (out, p) concatenated

  const size_t shmem = (size_t)2 * 64 * SROW * 2 + 64 * sizeof(float);  // ~258 KB (<320 KB WGP LDS)
  hipFuncSetAttribute(reinterpret_cast<const void*>(&attn_p2_wmma_kernel),
                      hipFuncAttributeMaxDynamicSharedMemorySize, (int)shmem);

  dim3 grid(8 * 12 * 16);   // B*H * (T/64)
  attn_p2_wmma_kernel<<<grid, dim3(128), shmem, stream>>>(q, k, v, am, out_o, p_o);
}
